// MLPBlock_85813446574554
// MI455X (gfx1250) — matrix operand order repaired and checked by reading, compiled, not run on hardware
//
#include <hip/hip_runtime.h>

#define E_ 16
#define TOPK 4
#define H_ 2048
#define I_ 2048
#define T_ 2048

#define BM 128
#define BN 64
#define BK 32
#define LDA 40   // halves per LDS row: 80B pitch = 16B-aligned for async B128,
                 // 20-dword stride -> conflict-free fragment reads

typedef __attribute__((ext_vector_type(16))) __bf16 v16bf;
typedef __attribute__((ext_vector_type(4)))  __bf16 v4bf;
typedef __attribute__((ext_vector_type(8)))  float  v8f;
typedef __attribute__((ext_vector_type(4)))  float  f4;

// ---- CDNA5 async global->LDS (ASYNCcnt-tracked, no VGPR round trip) -------
__device__ __forceinline__ void async_b128(unsigned lds_off,
                                           unsigned long long gaddr)
{
  asm volatile("global_load_async_to_lds_b128 %0, %1, off"
               :: "v"(lds_off), "v"(gaddr) : "memory");
}
__device__ __forceinline__ void wait_async()
{
  asm volatile("s_wait_asynccnt 0x0" ::: "memory");
}
__device__ __forceinline__ unsigned lds_off_of(const void* p)
{
  return (unsigned)(size_t)p;   // LDS offset lives in the low 32 bits
}

// ---------------------------------------------------------------------------
// Router: one wave32 per token.
// ---------------------------------------------------------------------------
__global__ __launch_bounds__(256) void moe_router(
    const float* __restrict__ x, const float* __restrict__ gw,
    const float* __restrict__ gb, int* __restrict__ tok_e,
    float* __restrict__ tok_w)
{
  const int wave = threadIdx.x >> 5;
  const int lane = threadIdx.x & 31;
  const int t = blockIdx.x * 8 + wave;
  if (t >= T_) return;

  float acc[E_];
#pragma unroll
  for (int e = 0; e < E_; ++e) acc[e] = 0.f;

  const float* xr = x + (size_t)t * H_;
  for (int h = lane; h < H_; h += 32) {
    const float xv = xr[h];
#pragma unroll
    for (int e = 0; e < E_; ++e) acc[e] += xv * gw[e * H_ + h];
  }
#pragma unroll
  for (int off = 16; off > 0; off >>= 1) {
#pragma unroll
    for (int e = 0; e < E_; ++e) acc[e] += __shfl_xor(acc[e], off, 32);
  }

  if (lane == 0) {
    float vals[E_];
#pragma unroll
    for (int e = 0; e < E_; ++e) vals[e] = acc[e] + gb[e];
    int idx[TOPK]; float tv[TOPK]; bool used[E_];
#pragma unroll
    for (int e = 0; e < E_; ++e) used[e] = false;
    for (int i = 0; i < TOPK; ++i) {     // strict > : ties pick lowest index
      float best = -3.0e38f; int bi = 0;
      for (int e = 0; e < E_; ++e)
        if (!used[e] && vals[e] > best) { best = vals[e]; bi = e; }
      used[bi] = true; idx[i] = bi; tv[i] = best;
    }
    const float m = tv[0];
    float w[TOPK]; float s = 0.f;
    for (int i = 0; i < TOPK; ++i) { w[i] = __expf(tv[i] - m); s += w[i]; }
    const float inv = 1.f / s;
    for (int i = 0; i < TOPK; ++i) {
      tok_e[t * TOPK + i] = idx[i];
      tok_w[t * TOPK + i] = w[i] * inv;
    }
  }
}

// ---------------------------------------------------------------------------
// Gather: deterministic compaction (no atomics), wave w = expert w.
// ---------------------------------------------------------------------------
__global__ __launch_bounds__(512) void moe_gather(
    const int* __restrict__ tok_e, const float* __restrict__ tok_w,
    int* __restrict__ ex_tok, float* __restrict__ ex_rw,
    int* __restrict__ ex_cnt, int* __restrict__ ex_off)
{
  __shared__ int laneCnt[E_][32];
  __shared__ int laneOff[E_][32];
  __shared__ int exCnt[E_], exOff[E_];
  const int e = threadIdx.x >> 5;
  const int lane = threadIdx.x & 31;
  const int t0 = lane * (T_ / 32);

  int cnt = 0;
  for (int i = 0; i < T_ / 32; ++i) {
    const int t = t0 + i;
#pragma unroll
    for (int j = 0; j < TOPK; ++j) cnt += (tok_e[t * TOPK + j] == e) ? 1 : 0;
  }
  laneCnt[e][lane] = cnt;
  __syncthreads();
  if (threadIdx.x < E_) {
    int s = 0;
    for (int l = 0; l < 32; ++l) { laneOff[threadIdx.x][l] = s; s += laneCnt[threadIdx.x][l]; }
    exCnt[threadIdx.x] = s;
  }
  __syncthreads();
  if (threadIdx.x == 0) {
    int s = 0;
    for (int k = 0; k < E_; ++k) { exOff[k] = s; s += exCnt[k]; }
  }
  __syncthreads();
  if (threadIdx.x < E_) {
    ex_cnt[threadIdx.x] = exCnt[threadIdx.x];
    ex_off[threadIdx.x] = exOff[threadIdx.x];
  }
  int w = exOff[e] + laneOff[e][lane];
  for (int i = 0; i < T_ / 32; ++i) {
    const int t = t0 + i;
#pragma unroll
    for (int j = 0; j < TOPK; ++j) {
      if (tok_e[t * TOPK + j] == e) {
        ex_tok[w] = t;
        ex_rw[w]  = tok_w[t * TOPK + j];
        ++w;
      }
    }
  }
}

__global__ void moe_zero(float* __restrict__ p, int n)
{
  const int i = blockIdx.x * blockDim.x + threadIdx.x;
  if (i < n) p[i] = 0.f;
}

// x fp32 -> bf16 (enables raw async A-copies in gemm1); 8 MB, one pass.
__global__ __launch_bounds__(256) void moe_cvt_x(
    const float* __restrict__ x, unsigned short* __restrict__ xb_raw, int n4)
{
  const int i = blockIdx.x * blockDim.x + threadIdx.x;
  if (i < n4) {
    const f4 v = ((const f4*)x)[i];
    v4bf o;
    o[0] = (__bf16)v.x; o[1] = (__bf16)v.y;
    o[2] = (__bf16)v.z; o[3] = (__bf16)v.w;
    ((v4bf*)xb_raw)[i] = o;
  }
}

// ---------------------------------------------------------------------------
// GEMM1 + SwiGLU (fused gate_up). A (gathered bf16 x rows) staged via
// global_load_async_to_lds_b128; B (w1 fp32) via load+cvt. Double-buffered.
// ---------------------------------------------------------------------------
__global__ __launch_bounds__(256) void moe_gemm1(
    const unsigned short* __restrict__ xb_raw, const float* __restrict__ w1,
    const float* __restrict__ b1, const float* __restrict__ alpha,
    const float* __restrict__ beta, const int* __restrict__ ex_tok,
    const int* __restrict__ ex_cnt, const int* __restrict__ ex_off,
    unsigned short* __restrict__ act_raw)
{
  const int e = blockIdx.z;
  const int cnt = ex_cnt[e];
  const int rtile = blockIdx.x;
  if (rtile * BM >= cnt) return;
  const int off = ex_off[e];
  const int j0 = blockIdx.y * BN;
  const __bf16* __restrict__ xb = (const __bf16*)xb_raw;
  __bf16* __restrict__ act = (__bf16*)act_raw;

  __shared__ __bf16 As[2][BM][LDA];
  __shared__ __bf16 Bg[2][BN][LDA];
  __shared__ __bf16 Bl[2][BN][LDA];

  const int tid  = threadIdx.x;
  const int lane = tid & 31;
  const int wave = tid >> 5;
  const int wm = wave >> 1;
  const int wn = wave & 1;
  const int nlo = lane & 15;
  const int hi  = lane >> 4;

  // A staging: 2 threads per row, each issues 2 async B128 (32B)
  const int arow = tid >> 1;
  const int aseg = tid & 1;
  const int g_r  = rtile * BM + arow;
  const bool av  = (g_r < cnt);
  const int tokA = av ? ex_tok[off + g_r] : 0;
  const unsigned long long xa =
      (unsigned long long)(size_t)(xb + (size_t)tokA * H_ + aseg * 16);

  // B staging: 4 threads per row, 8 fp32 each
  const int brow = tid >> 2;
  const int bcb  = (tid & 3) * 8;
  const float* __restrict__ w1e = w1 + (size_t)e * (2 * I_) * H_;
  const float* __restrict__ bgs = w1e + (size_t)(j0 + brow) * H_;
  const float* __restrict__ bls = w1e + (size_t)(I_ + j0 + brow) * H_;

  const v8f vzero = {0.f, 0.f, 0.f, 0.f, 0.f, 0.f, 0.f, 0.f};
  v8f accg[2][2], accl[2][2];
#pragma unroll
  for (int mt = 0; mt < 2; ++mt)
#pragma unroll
    for (int nt = 0; nt < 2; ++nt) { accg[mt][nt] = vzero; accl[mt][nt] = vzero; }

  auto stage = [&](int buf, int kk) {
    if (av) {
      const unsigned l0 = lds_off_of(&As[buf][arow][aseg * 16]);
      const unsigned long long g0 = xa + (unsigned long long)(kk * 2);
      async_b128(l0, g0);
      async_b128(l0 + 16u, g0 + 16ull);
    }
    const f4 v0 = *(const f4*)(bgs + kk + bcb);
    const f4 v1 = *(const f4*)(bgs + kk + bcb + 4);
    const f4 u0 = *(const f4*)(bls + kk + bcb);
    const f4 u1 = *(const f4*)(bls + kk + bcb + 4);
    Bg[buf][brow][bcb + 0] = (__bf16)v0.x; Bg[buf][brow][bcb + 1] = (__bf16)v0.y;
    Bg[buf][brow][bcb + 2] = (__bf16)v0.z; Bg[buf][brow][bcb + 3] = (__bf16)v0.w;
    Bg[buf][brow][bcb + 4] = (__bf16)v1.x; Bg[buf][brow][bcb + 5] = (__bf16)v1.y;
    Bg[buf][brow][bcb + 6] = (__bf16)v1.z; Bg[buf][brow][bcb + 7] = (__bf16)v1.w;
    Bl[buf][brow][bcb + 0] = (__bf16)u0.x; Bl[buf][brow][bcb + 1] = (__bf16)u0.y;
    Bl[buf][brow][bcb + 2] = (__bf16)u0.z; Bl[buf][brow][bcb + 3] = (__bf16)u0.w;
    Bl[buf][brow][bcb + 4] = (__bf16)u1.x; Bl[buf][brow][bcb + 5] = (__bf16)u1.y;
    Bl[buf][brow][bcb + 6] = (__bf16)u1.z; Bl[buf][brow][bcb + 7] = (__bf16)u1.w;
  };

  auto compute = [&](int buf) {
    v16bf af[2], bgf[2], blf[2];
#pragma unroll
    for (int mt = 0; mt < 2; ++mt) {
      const int r = wm * 32 + mt * 16 + nlo;         // M = lane&15
#pragma unroll
      for (int v = 0; v < 8; ++v) {
        const int k0 = ((v & 3) << 1) + (hi << 3) + ((v >> 2) << 4);
        af[mt][2 * v]     = As[buf][r][k0];
        af[mt][2 * v + 1] = As[buf][r][k0 + 1];
      }
    }
#pragma unroll
    for (int nt = 0; nt < 2; ++nt) {
      const int c = wn * 32 + nt * 16 + nlo;         // N = lane&15
#pragma unroll
      for (int v = 0; v < 8; ++v) {
        const int k0 = ((v & 3) << 1) + (hi << 3) + ((v >> 2) << 4);
        bgf[nt][2 * v]     = Bg[buf][c][k0];
        bgf[nt][2 * v + 1] = Bg[buf][c][k0 + 1];
        blf[nt][2 * v]     = Bl[buf][c][k0];
        blf[nt][2 * v + 1] = Bl[buf][c][k0 + 1];
      }
    }
#pragma unroll
    for (int mt = 0; mt < 2; ++mt)
#pragma unroll
      for (int nt = 0; nt < 2; ++nt) {
        accg[mt][nt] = __builtin_amdgcn_wmma_f32_16x16x32_bf16(
            false, af[mt], false, bgf[nt], (short)0, accg[mt][nt], false, false);
        accl[mt][nt] = __builtin_amdgcn_wmma_f32_16x16x32_bf16(
            false, af[mt], false, blf[nt], (short)0, accl[mt][nt], false, false);
      }
  };

  stage(0, 0);
  wait_async();
  __syncthreads();
  int buf = 0;
  for (int kk = 0; kk < H_; kk += BK) {
    if (kk + BK < H_) stage(buf ^ 1, kk + BK);
    compute(buf);
    wait_async();
    __syncthreads();
    buf ^= 1;
  }

  const float aE = alpha[e];
  const float bE = beta[e];
  const float* __restrict__ b1e = b1 + (size_t)e * (2 * I_);
#pragma unroll
  for (int nt = 0; nt < 2; ++nt) {
    const int jc = j0 + wn * 32 + nt * 16 + nlo;
    const float b1g = b1e[jc];
    const float b1l = b1e[I_ + jc];
#pragma unroll
    for (int mt = 0; mt < 2; ++mt) {
#pragma unroll
      for (int v = 0; v < 8; ++v) {
        const int gr = rtile * BM + wm * 32 + mt * 16 + v + hi * 8;
        if (gr < cnt) {
          const float g = accg[mt][nt][v] + b1g;
          const float l = accl[mt][nt][v] + b1l;
          const float sg = 1.f / (1.f + __expf(-aE * g));
          act[(size_t)(off + gr) * I_ + jc] = (__bf16)(g * sg * (l + bE));
        }
      }
    }
  }
}

// ---------------------------------------------------------------------------
// GEMM2: y = act @ w2_e.T + b2 ; out[token] += rw * y. Launched per expert,
// sequentially -> race-free, deterministic accumulation order (0..15).
// A (bf16 act) staged async; B (w2 fp32) via load+cvt. Double-buffered.
// ---------------------------------------------------------------------------
__global__ __launch_bounds__(256) void moe_gemm2(
    const unsigned short* __restrict__ act_raw, const float* __restrict__ w2,
    const float* __restrict__ b2, const int* __restrict__ ex_tok,
    const float* __restrict__ ex_rw, const int* __restrict__ ex_cnt,
    const int* __restrict__ ex_off, float* __restrict__ out, int e)
{
  const int cnt = ex_cnt[e];
  const int rtile = blockIdx.x;
  if (rtile * BM >= cnt) return;
  const int off = ex_off[e];
  const int c0 = blockIdx.y * BN;
  const __bf16* __restrict__ act = (const __bf16*)act_raw;

  __shared__ __bf16 As[2][BM][LDA];
  __shared__ __bf16 Bs[2][BN][LDA];

  const int tid  = threadIdx.x;
  const int lane = tid & 31;
  const int wave = tid >> 5;
  const int wm = wave >> 1;
  const int wn = wave & 1;
  const int nlo = lane & 15;
  const int hi  = lane >> 4;

  const int arow = tid >> 1;
  const int aseg = tid & 1;
  const int g_r  = rtile * BM + arow;
  const bool av  = (g_r < cnt);
  const unsigned long long aa = (unsigned long long)(size_t)(
      act + (size_t)(off + (av ? g_r : 0)) * I_ + aseg * 16);

  const int brow = tid >> 2;
  const int bcb  = (tid & 3) * 8;
  const float* __restrict__ w2e = w2 + (size_t)e * H_ * I_;
  const float* __restrict__ bsr = w2e + (size_t)(c0 + brow) * I_;

  const v8f vzero = {0.f, 0.f, 0.f, 0.f, 0.f, 0.f, 0.f, 0.f};
  v8f acc[2][2];
#pragma unroll
  for (int mt = 0; mt < 2; ++mt)
#pragma unroll
    for (int nt = 0; nt < 2; ++nt) acc[mt][nt] = vzero;

  auto stage = [&](int buf, int kk) {
    if (av) {
      const unsigned l0 = lds_off_of(&As[buf][arow][aseg * 16]);
      const unsigned long long g0 = aa + (unsigned long long)(kk * 2);
      async_b128(l0, g0);
      async_b128(l0 + 16u, g0 + 16ull);
    }
    const f4 v0 = *(const f4*)(bsr + kk + bcb);
    const f4 v1 = *(const f4*)(bsr + kk + bcb + 4);
    Bs[buf][brow][bcb + 0] = (__bf16)v0.x; Bs[buf][brow][bcb + 1] = (__bf16)v0.y;
    Bs[buf][brow][bcb + 2] = (__bf16)v0.z; Bs[buf][brow][bcb + 3] = (__bf16)v0.w;
    Bs[buf][brow][bcb + 4] = (__bf16)v1.x; Bs[buf][brow][bcb + 5] = (__bf16)v1.y;
    Bs[buf][brow][bcb + 6] = (__bf16)v1.z; Bs[buf][brow][bcb + 7] = (__bf16)v1.w;
  };

  auto compute = [&](int buf) {
    v16bf af[2], bf_[2];
#pragma unroll
    for (int mt = 0; mt < 2; ++mt) {
      const int r = wm * 32 + mt * 16 + nlo;
#pragma unroll
      for (int v = 0; v < 8; ++v) {
        const int k0 = ((v & 3) << 1) + (hi << 3) + ((v >> 2) << 4);
        af[mt][2 * v]     = As[buf][r][k0];
        af[mt][2 * v + 1] = As[buf][r][k0 + 1];
      }
    }
#pragma unroll
    for (int nt = 0; nt < 2; ++nt) {
      const int c = wn * 32 + nt * 16 + nlo;
#pragma unroll
      for (int v = 0; v < 8; ++v) {
        const int k0 = ((v & 3) << 1) + (hi << 3) + ((v >> 2) << 4);
        bf_[nt][2 * v]     = Bs[buf][c][k0];
        bf_[nt][2 * v + 1] = Bs[buf][c][k0 + 1];
      }
    }
#pragma unroll
    for (int mt = 0; mt < 2; ++mt)
#pragma unroll
      for (int nt = 0; nt < 2; ++nt)
        acc[mt][nt] = __builtin_amdgcn_wmma_f32_16x16x32_bf16(
            false, af[mt], false, bf_[nt], (short)0, acc[mt][nt], false, false);
  };

  stage(0, 0);
  wait_async();
  __syncthreads();
  int buf = 0;
  for (int kk = 0; kk < I_; kk += BK) {
    if (kk + BK < I_) stage(buf ^ 1, kk + BK);
    compute(buf);
    wait_async();
    __syncthreads();
    buf ^= 1;
  }

  const float* __restrict__ b2e = b2 + (size_t)e * H_;
#pragma unroll
  for (int mt = 0; mt < 2; ++mt) {
    int toks[8]; float rws[8]; bool valid[8];
#pragma unroll
    for (int v = 0; v < 8; ++v) {
      const int gr = rtile * BM + wm * 32 + mt * 16 + v + hi * 8;
      valid[v] = (gr < cnt);
      toks[v] = valid[v] ? ex_tok[off + gr] : 0;
      rws[v]  = valid[v] ? ex_rw[off + gr] : 0.f;
    }
#pragma unroll
    for (int nt = 0; nt < 2; ++nt) {
      const int cc = c0 + wn * 32 + nt * 16 + nlo;
      const float bb = b2e[cc];
#pragma unroll
      for (int v = 0; v < 8; ++v) {
        if (valid[v]) {
          float* dst = out + (size_t)toks[v] * H_ + cc;
          *dst += rws[v] * (acc[mt][nt][v] + bb);
        }
      }
    }
  }
}

// ---------------------------------------------------------------------------
extern "C" void kernel_launch(void* const* d_in, const int* in_sizes, int n_in,
                              void* d_out, int out_size, void* d_ws, size_t ws_size,
                              hipStream_t stream)
{
  const float* x  = (const float*)d_in[0];
  const float* gw = (const float*)d_in[1];
  const float* gb = (const float*)d_in[2];
  const float* w1 = (const float*)d_in[3];
  const float* b1 = (const float*)d_in[4];
  const float* w2 = (const float*)d_in[5];
  const float* b2 = (const float*)d_in[6];
  const float* al = (const float*)d_in[7];
  const float* be = (const float*)d_in[8];
  float* out = (float*)d_out;

  // workspace layout (~42 MB)
  char* ws = (char*)d_ws;
  unsigned short* act = (unsigned short*)ws; ws += (size_t)T_ * TOPK * I_ * 2;
  unsigned short* xb  = (unsigned short*)ws; ws += (size_t)T_ * H_ * 2;
  int*   tok_e  = (int*)ws;   ws += (size_t)T_ * TOPK * 4;
  float* tok_w  = (float*)ws; ws += (size_t)T_ * TOPK * 4;
  int*   ex_tok = (int*)ws;   ws += (size_t)T_ * TOPK * 4;
  float* ex_rw  = (float*)ws; ws += (size_t)T_ * TOPK * 4;
  int*   ex_cnt = (int*)ws;   ws += 64;
  int*   ex_off = (int*)ws;   ws += 64;

  moe_cvt_x<<<(T_ * H_ / 4 + 255) / 256, 256, 0, stream>>>(x, xb, T_ * H_ / 4);
  moe_router<<<T_ / 8, 256, 0, stream>>>(x, gw, gb, tok_e, tok_w);
  moe_gather<<<1, 512, 0, stream>>>(tok_e, tok_w, ex_tok, ex_rw, ex_cnt, ex_off);
  moe_zero<<<(T_ * H_ + 255) / 256, 256, 0, stream>>>(out, T_ * H_);

  dim3 g1(T_ / BM, I_ / BN, E_);           // worst-case row tiles; extras exit
  moe_gemm1<<<g1, 256, 0, stream>>>(xb, w1, b1, al, be, ex_tok, ex_cnt, ex_off, act);

  dim3 g2(T_ / BM, H_ / BN);
  for (int e = 0; e < E_; ++e)             // sequential: deterministic accumulation
    moe_gemm2<<<g2, 256, 0, stream>>>(act, w2, b2, ex_tok, ex_rw, ex_cnt, ex_off, out, e);
}
